// Informer_20186346291968
// MI455X (gfx1250) — matrix operand order repaired and checked by reading, not compiled, not run on hardware
//
#include <hip/hip_runtime.h>
#include <math.h>

// ---------------------------------------------------------------------------
// CDNA5 (gfx1250) wave32 WMMA types / helpers
// ---------------------------------------------------------------------------
typedef __attribute__((ext_vector_type(16))) __bf16 v16bf;
typedef __attribute__((ext_vector_type(8)))  float  v8f;

__device__ __forceinline__ unsigned short f2bf(float f) {
  unsigned int u = __builtin_bit_cast(unsigned int, f);
  u += 0x7FFFu + ((u >> 16) & 1u);            // round-to-nearest-even
  return (unsigned short)(u >> 16);
}

struct __align__(8) us4 { unsigned short x, y, z, w; };
__device__ __forceinline__ us4 cvt4(float4 v) {
  us4 u; u.x = f2bf(v.x); u.y = f2bf(v.y); u.z = f2bf(v.z); u.w = f2bf(v.w);
  return u;
}

__device__ __forceinline__ v8f wmma_bf16(v16bf a, v16bf b, v8f c) {
  return __builtin_amdgcn_wmma_f32_16x16x32_bf16(false, a, false, b, (short)0, c,
                                                 false, false);
}

// Fragment loaders: tiles stored in LDS in fragment-native order so each
// fragment is two contiguous 16B chunks -> two ds_load_b128 per fragment.
struct __align__(16) B32x2 { uint4 a, b; };

// A-matrix (16xK bf16, row-major rows in LDS).  Lane16 = row; halves:
// k = fbase + (i&7) + (i&8?16:0) + hi*8  -> chunks at bytes [f+hi*16], [f+32+hi*16]
__device__ __forceinline__ v16bf fragA(const unsigned short* rowp, int fbyte, int hi) {
  const char* p = (const char*)rowp + fbyte + hi * 16;
  B32x2 t;
  t.a = *(const uint4*)p;
  t.b = *(const uint4*)(p + 32);
  return __builtin_bit_cast(v16bf, t);
}
// B-matrix (Kx16 bf16, stored column-major in LDS).  Lane16 = col; halves:
// k = fbase + (i&7) + (i&8?16:0) + hi*8  -> chunks at bytes [f+hi*16], [f+32+hi*16]
__device__ __forceinline__ v16bf fragB(const unsigned short* colp, int fbyte, int hi) {
  const char* p = (const char*)colp + fbyte + hi * 16;
  B32x2 t;
  t.a = *(const uint4*)p;
  t.b = *(const uint4*)(p + 32);
  return __builtin_bit_cast(v16bf, t);
}

__device__ __forceinline__ float gelu_f(float x) {
  return 0.5f * x * (1.0f + erff(x * 0.7071067811865476f));
}

// ---------------------------------------------------------------------------
// Generic GEMM: C[M,N] = act(A[M,K] @ B[K,N] + bias), f32 in/out, bf16 WMMA.
// Block tile 64x128, 8 waves (4x2), each wave 16x64 -> 4 WMMA per K-step.
// K must be a multiple of 32 (true for all uses here).
// ---------------------------------------------------------------------------
#define GTM 64
#define GTN 128
#define GTK 32
#define ALD 40   // padded row stride (halves) for As
#define BLD 40   // padded col stride (halves) for Bs (transposed)

__global__ __launch_bounds__(256) void gemm_bf16_kernel(
    const float* __restrict__ A, const float* __restrict__ B,
    const float* __restrict__ bias, float* __restrict__ C,
    int M, int N, int K, int act) {
  __shared__ __align__(16) unsigned short As[GTM * ALD];   // row-major, padded
  __shared__ __align__(16) unsigned short Bs[GTN * BLD];   // column-major, padded
  const int m0 = blockIdx.x * GTM;
  const int n0 = blockIdx.y * GTN;
  const int tid = threadIdx.x;
  const int wave = tid >> 5;
  const int lane = tid & 31;
  const int lane16 = lane & 15, hi = lane >> 4;
  const int wm = wave >> 1, wn = wave & 1;           // 4x2 wave grid
  const bool fullA = (m0 + GTM <= M);
  const bool fullB = (n0 + GTN <= N);
  v8f acc[4] = {{}, {}, {}, {}};

  for (int k0 = 0; k0 < K; k0 += GTK) {
    if (fullA) {                                     // 64x32 f32 = 512 float4
#pragma unroll
      for (int j = 0; j < 2; ++j) {
        int fid = tid + j * 256;
        int row = fid >> 3, c4 = fid & 7;
        float4 v = *(const float4*)(A + (size_t)(m0 + row) * K + k0 + c4 * 4);
        *(us4*)(&As[row * ALD + c4 * 4]) = cvt4(v);
      }
    } else {
      for (int i = tid; i < GTM * GTK; i += 256) {
        int r = i >> 5, c = i & 31;
        int gr = m0 + r;
        float v = (gr < M) ? A[(size_t)gr * K + k0 + c] : 0.f;
        As[r * ALD + c] = f2bf(v);
      }
    }
    if (fullB) {                                     // 32x128 f32 = 1024 float4
#pragma unroll
      for (int j = 0; j < 4; ++j) {
        int fid = tid + j * 256;
        int row = fid >> 5, c4 = fid & 31;
        float4 v = *(const float4*)(B + (size_t)(k0 + row) * N + n0 + c4 * 4);
        int c = c4 * 4;
        Bs[(c + 0) * BLD + row] = f2bf(v.x);
        Bs[(c + 1) * BLD + row] = f2bf(v.y);
        Bs[(c + 2) * BLD + row] = f2bf(v.z);
        Bs[(c + 3) * BLD + row] = f2bf(v.w);
      }
    } else {
      for (int i = tid; i < GTK * GTN; i += 256) {
        int r = i >> 7, c = i & 127;
        int gc = n0 + c;
        float v = (gc < N) ? B[(size_t)(k0 + r) * N + gc] : 0.f;
        Bs[c * BLD + r] = f2bf(v);
      }
    }
    __syncthreads();
    v16bf a = fragA(&As[(wm * 16 + lane16) * ALD], 0, hi);
#pragma unroll
    for (int t = 0; t < 4; ++t) {
      v16bf b = fragB(&Bs[(wn * 64 + t * 16 + lane16) * BLD], 0, hi);
      acc[t] = wmma_bf16(a, b, acc[t]);
    }
    __syncthreads();
  }
#pragma unroll
  for (int t = 0; t < 4; ++t) {
#pragma unroll
    for (int r = 0; r < 8; ++r) {
      int row = m0 + wm * 16 + r + hi * 8;
      int col = n0 + wn * 64 + t * 16 + lane16;
      if (row < M && col < N) {
        float v = acc[t][r];
        if (bias) v += bias[col];
        if (act == 1) v = gelu_f(v);
        C[(size_t)row * N + col] = v;
      }
    }
  }
}

// ---------------------------------------------------------------------------
// LayerNorm over last dim D (optional fused residual): O = LN(X [+ R])
// ---------------------------------------------------------------------------
__global__ __launch_bounds__(256) void ln_kernel(
    const float* __restrict__ X, const float* __restrict__ R,
    const float* __restrict__ g, const float* __restrict__ b,
    float* __restrict__ O, int D) {
  const int row = blockIdx.x;
  const float* x = X + (size_t)row * D;
  const float* r = R ? R + (size_t)row * D : nullptr;
  float* o = O + (size_t)row * D;
  __shared__ float red[256];
  float s = 0.f;
  for (int i = threadIdx.x; i < D; i += 256) s += x[i] + (r ? r[i] : 0.f);
  red[threadIdx.x] = s; __syncthreads();
  for (int st = 128; st > 0; st >>= 1) {
    if (threadIdx.x < st) red[threadIdx.x] += red[threadIdx.x + st];
    __syncthreads();
  }
  float mu = red[0] / D; __syncthreads();
  float vs = 0.f;
  for (int i = threadIdx.x; i < D; i += 256) {
    float v = x[i] + (r ? r[i] : 0.f) - mu; vs += v * v;
  }
  red[threadIdx.x] = vs; __syncthreads();
  for (int st = 128; st > 0; st >>= 1) {
    if (threadIdx.x < st) red[threadIdx.x] += red[threadIdx.x + st];
    __syncthreads();
  }
  float inv = rsqrtf(red[0] / D + 1e-5f);
  for (int i = threadIdx.x; i < D; i += 256) {
    float v = x[i] + (r ? r[i] : 0.f);
    o[i] = (v - mu) * inv * g[i] + b[i];
  }
}

// ---------------------------------------------------------------------------
// ProbSparse attention.  QKV layout: (B, n, 1536) part-major [3][h=8][dk=64].
// All 8 waves of a block share (b,h) (tiles-per-bh is a multiple of 8), so K/V
// are staged block-cooperatively; Q and P tiles are per-wave.
// ---------------------------------------------------------------------------

// Phase A: m[b,h,row] = max_k - mean_k of scaled scores.
__global__ __launch_bounds__(256) void attn_m_kernel(
    const float* __restrict__ QKV, float* __restrict__ Mout, int n) {
  __shared__ __align__(16) unsigned short Qs[8][16 * 64];   // per-wave Q tile
  __shared__ __align__(16) unsigned short Kc[16 * 64];      // shared K tile
  const int tid = threadIdx.x;
  const int wave = tid >> 5;
  const int lane = tid & 31;
  const int lane16 = lane & 15, hi = lane >> 4;
  const int ntiles = n >> 4;
  const int bh = (blockIdx.x * 8) / ntiles;
  const int qt = (blockIdx.x * 8) % ntiles + wave;
  const int b = bh >> 3, h = bh & 7;
  const size_t rs = 1536;
  const float* base = QKV + (size_t)b * n * rs + (size_t)h * 64;

  // stage own Q tile (16x64) as bf16, row-major
#pragma unroll
  for (int j = 0; j < 8; ++j) {
    int fid = j * 32 + lane;
    int row = fid >> 4, c4 = fid & 15;
    float4 v = *(const float4*)(base + (size_t)(qt * 16 + row) * rs + c4 * 4);
    *(us4*)(&Qs[wave][row * 64 + c4 * 4]) = cvt4(v);
  }
  v16bf aq0 = fragA(&Qs[wave][lane16 * 64], 0, hi);
  v16bf aq1 = fragA(&Qs[wave][lane16 * 64], 64, hi);

  float rmax[8], rsum[8];
#pragma unroll
  for (int r = 0; r < 8; ++r) { rmax[r] = -3.4e38f; rsum[r] = 0.f; }

  for (int ct = 0; ct < ntiles; ++ct) {
    __syncthreads();
    {   // 16 rows x 16 float4 = 256 -> one per thread
      int row = tid >> 4, c4 = tid & 15;
      float4 v = *(const float4*)(base + (size_t)(ct * 16 + row) * rs + 512 + c4 * 4);
      *(us4*)(&Kc[row * 64 + c4 * 4]) = cvt4(v);
    }
    __syncthreads();
    v16bf bk0 = fragB(&Kc[lane16 * 64], 0, hi);
    v16bf bk1 = fragB(&Kc[lane16 * 64], 64, hi);
    v8f s = {};
    s = wmma_bf16(aq0, bk0, s);
    s = wmma_bf16(aq1, bk1, s);
#pragma unroll
    for (int r = 0; r < 8; ++r) {
      float v = s[r] * 0.125f;
      rmax[r] = fmaxf(rmax[r], v);
      rsum[r] += v;
    }
  }
#pragma unroll
  for (int r = 0; r < 8; ++r) {
#pragma unroll
    for (int mk = 1; mk < 16; mk <<= 1) {
      rmax[r] = fmaxf(rmax[r], __shfl_xor(rmax[r], mk, 32));
      rsum[r] += __shfl_xor(rsum[r], mk, 32);
    }
  }
  if (lane16 == 0) {
#pragma unroll
    for (int r = 0; r < 8; ++r) {
      int row = qt * 16 + r + hi * 8;
      Mout[(size_t)bh * n + row] = rmax[r] - rsum[r] / n;
    }
  }
}

// Phase B: per-(b,h) threshold tau ~= n_top-th largest of m (binary search).
__global__ __launch_bounds__(256) void topk_tau_kernel(
    const float* __restrict__ Mv, float* __restrict__ Tau, int n, int ntop) {
  const int bh = blockIdx.x;
  __shared__ float sm[1024];
  __shared__ float red[256];
  const float* m = Mv + (size_t)bh * n;
  for (int i = threadIdx.x; i < n; i += 256) sm[i] = m[i];
  __syncthreads();
  float lmin = 3.4e38f, lmax = -3.4e38f;
  for (int i = threadIdx.x; i < n; i += 256) {
    lmin = fminf(lmin, sm[i]); lmax = fmaxf(lmax, sm[i]);
  }
  red[threadIdx.x] = lmin; __syncthreads();
  for (int st = 128; st > 0; st >>= 1) {
    if (threadIdx.x < st) red[threadIdx.x] = fminf(red[threadIdx.x], red[threadIdx.x + st]);
    __syncthreads();
  }
  float lo = red[0]; __syncthreads();
  red[threadIdx.x] = lmax; __syncthreads();
  for (int st = 128; st > 0; st >>= 1) {
    if (threadIdx.x < st) red[threadIdx.x] = fmaxf(red[threadIdx.x], red[threadIdx.x + st]);
    __syncthreads();
  }
  float hiv = red[0]; __syncthreads();
  for (int it = 0; it < 30; ++it) {
    float mid = 0.5f * (lo + hiv);
    int c = 0;
    for (int i = threadIdx.x; i < n; i += 256) c += (sm[i] > mid) ? 1 : 0;
    red[threadIdx.x] = (float)c; __syncthreads();
    for (int st = 128; st > 0; st >>= 1) {
      if (threadIdx.x < st) red[threadIdx.x] += red[threadIdx.x + st];
      __syncthreads();
    }
    int cnt = (int)red[0]; __syncthreads();
    if (cnt >= ntop) lo = mid; else hiv = mid;
  }
  if (threadIdx.x == 0) Tau[bh] = lo;
}

// Phase C: flash-style online-softmax attention with ProbSparse row masking.
// Out layout: (B, n, 512) head-major (== transpose(0,2,1,3).reshape).
__global__ __launch_bounds__(256) void attn_out_kernel(
    const float* __restrict__ QKV, const float* __restrict__ Mv,
    const float* __restrict__ Tau, float* __restrict__ Out, int n) {
  __shared__ __align__(16) unsigned short Qs[8][16 * 64];  // per-wave Q
  __shared__ __align__(16) unsigned short Kc[32 * 64];     // shared K chunk (row-major)
  __shared__ __align__(16) unsigned short Vt[64 * 32];     // shared V chunk (transposed)
  __shared__ __align__(16) unsigned short Pt[8][16 * 32];  // per-wave P tile
  const int tid = threadIdx.x;
  const int wave = tid >> 5;
  const int lane = tid & 31;
  const int lane16 = lane & 15, hi = lane >> 4;
  const int ntiles = n >> 4;
  const int bh = (blockIdx.x * 8) / ntiles;
  const int qt = (blockIdx.x * 8) % ntiles + wave;
  const int b = bh >> 3, h = bh & 7;
  const size_t rs = 1536;
  const float* base = QKV + (size_t)b * n * rs + (size_t)h * 64;
  const float tau = Tau[bh];

#pragma unroll
  for (int j = 0; j < 8; ++j) {
    int fid = j * 32 + lane;
    int row = fid >> 4, c4 = fid & 15;
    float4 v = *(const float4*)(base + (size_t)(qt * 16 + row) * rs + c4 * 4);
    *(us4*)(&Qs[wave][row * 64 + c4 * 4]) = cvt4(v);
  }
  v16bf aq0 = fragA(&Qs[wave][lane16 * 64], 0, hi);
  v16bf aq1 = fragA(&Qs[wave][lane16 * 64], 64, hi);

  bool sel[8];
#pragma unroll
  for (int r = 0; r < 8; ++r) {
    int row = qt * 16 + r + hi * 8;
    sel[r] = (Mv[(size_t)bh * n + row] >= tau);
  }
  float Mr[8], Sr[8];
  v8f O0 = {}, O1 = {}, O2 = {}, O3 = {};
#pragma unroll
  for (int r = 0; r < 8; ++r) { Mr[r] = -3.4e38f; Sr[r] = 0.f; }

  const int nchunks = n >> 5;
  for (int c = 0; c < nchunks; ++c) {
    __syncthreads();
#pragma unroll
    for (int j = 0; j < 2; ++j) {       // 32 rows x 16 float4 (K and V each)
      int fid = tid + j * 256;
      int row = fid >> 4, c4 = fid & 15;
      const float* kr = base + (size_t)(c * 32 + row) * rs + 512;
      *(us4*)(&Kc[row * 64 + c4 * 4]) = cvt4(*(const float4*)(kr + c4 * 4));
      const float* vr = base + (size_t)(c * 32 + row) * rs + 1024;
      float4 vv = *(const float4*)(vr + c4 * 4);
      int cc = c4 * 4;
      Vt[(cc + 0) * 32 + row] = f2bf(vv.x);
      Vt[(cc + 1) * 32 + row] = f2bf(vv.y);
      Vt[(cc + 2) * 32 + row] = f2bf(vv.z);
      Vt[(cc + 3) * 32 + row] = f2bf(vv.w);
    }
    __syncthreads();
    v16bf bkA0 = fragB(&Kc[lane16 * 64], 0, hi);
    v16bf bkA1 = fragB(&Kc[lane16 * 64], 64, hi);
    v16bf bkB0 = fragB(&Kc[(16 + lane16) * 64], 0, hi);
    v16bf bkB1 = fragB(&Kc[(16 + lane16) * 64], 64, hi);
    v8f s0 = {}, s1 = {};
    s0 = wmma_bf16(aq0, bkA0, s0); s0 = wmma_bf16(aq1, bkA1, s0);
    s1 = wmma_bf16(aq0, bkB0, s1); s1 = wmma_bf16(aq1, bkB1, s1);

#pragma unroll
    for (int r = 0; r < 8; ++r) {
      float v0 = sel[r] ? s0[r] * 0.125f : -1e9f;
      float v1 = sel[r] ? s1[r] * 0.125f : -1e9f;
      float t = fmaxf(v0, v1);
#pragma unroll
      for (int mk = 1; mk < 16; mk <<= 1) t = fmaxf(t, __shfl_xor(t, mk, 32));
      float nm = fmaxf(Mr[r], t);
      float fr = __expf(Mr[r] - nm);          // first chunk: exp(-inf)=0
      float e0 = __expf(v0 - nm), e1 = __expf(v1 - nm);
      float cs = e0 + e1;
#pragma unroll
      for (int mk = 1; mk < 16; mk <<= 1) cs += __shfl_xor(cs, mk, 32);
      Sr[r] = Sr[r] * fr + cs;
      Mr[r] = nm;
      O0[r] *= fr; O1[r] *= fr; O2[r] *= fr; O3[r] *= fr;
      Pt[wave][(r + hi * 8) * 32 + lane16]      = f2bf(e0);
      Pt[wave][(r + hi * 8) * 32 + 16 + lane16] = f2bf(e1);
    }
    // intra-wave LDS RAW; compiler inserts s_wait_dscnt
    v16bf pa  = fragA(&Pt[wave][lane16 * 32], 0, hi);
    v16bf bv0 = fragB(&Vt[( 0 + lane16) * 32], 0, hi);
    v16bf bv1 = fragB(&Vt[(16 + lane16) * 32], 0, hi);
    v16bf bv2 = fragB(&Vt[(32 + lane16) * 32], 0, hi);
    v16bf bv3 = fragB(&Vt[(48 + lane16) * 32], 0, hi);
    O0 = wmma_bf16(pa, bv0, O0);
    O1 = wmma_bf16(pa, bv1, O1);
    O2 = wmma_bf16(pa, bv2, O2);
    O3 = wmma_bf16(pa, bv3, O3);
  }
#pragma unroll
  for (int r = 0; r < 8; ++r) {
    int row = qt * 16 + r + hi * 8;
    float inv = 1.f / Sr[r];
    float* orow = Out + ((size_t)(b * n + row)) * 512 + h * 64;
    orow[ 0 + lane16] = O0[r] * inv;
    orow[16 + lane16] = O1[r] * inv;
    orow[32 + lane16] = O2[r] * inv;
    orow[48 + lane16] = O3[r] * inv;
  }
}

// ---------------------------------------------------------------------------
// Conv-distil support
// ---------------------------------------------------------------------------
__global__ __launch_bounds__(256) void conv_pack_kernel(
    const float* __restrict__ W, float* __restrict__ P) {
  int i = blockIdx.x * 256 + threadIdx.x;
  if (i >= 1536 * 512) return;
  int kk = i / 512, co = i % 512;
  int k = kk / 512, ci = kk % 512;
  P[i] = W[(size_t)co * 1536 + ci * 3 + k];
}

__global__ __launch_bounds__(256) void im2col_kernel(
    const float* __restrict__ X, float* __restrict__ Y, int n) {
  size_t total = (size_t)16 * n * 1536;
  size_t i = (size_t)blockIdx.x * 256 + threadIdx.x;
  if (i >= total) return;
  int ci = (int)(i % 512);
  size_t rem = i / 512;
  int k = (int)(rem % 3);
  size_t bt = rem / 3;
  int t = (int)(bt % n), b = (int)(bt / n);
  int ts = t + k - 1;
  if (ts < 0) ts += n; else if (ts >= n) ts -= n;
  Y[i] = X[((size_t)b * n + ts) * 512 + ci];
}

__global__ __launch_bounds__(256) void pool_elu_ln_kernel(
    const float* __restrict__ X, const float* __restrict__ g,
    const float* __restrict__ bb, float* __restrict__ O, int n) {
  const int rowo = blockIdx.x;
  const int n2 = n >> 1;
  const int b = rowo / n2, t2 = rowo % n2;
  __shared__ float buf[512];
  __shared__ float red[256];
  for (int c = threadIdx.x; c < 512; c += 256) {
    float v = -3.4e38f;
    for (int dt = -1; dt <= 1; ++dt) {
      int t = 2 * t2 + dt;
      if (t >= 0 && t < n) v = fmaxf(v, X[((size_t)b * n + t) * 512 + c]);
    }
    buf[c] = (v > 0.f) ? v : (__expf(v) - 1.f);   // ELU
  }
  __syncthreads();
  float s = 0.f;
  for (int c = threadIdx.x; c < 512; c += 256) s += buf[c];
  red[threadIdx.x] = s; __syncthreads();
  for (int st = 128; st > 0; st >>= 1) {
    if (threadIdx.x < st) red[threadIdx.x] += red[threadIdx.x + st];
    __syncthreads();
  }
  float mu = red[0] / 512.f; __syncthreads();
  float vs = 0.f;
  for (int c = threadIdx.x; c < 512; c += 256) {
    float v = buf[c] - mu; vs += v * v;
  }
  red[threadIdx.x] = vs; __syncthreads();
  for (int st = 128; st > 0; st >>= 1) {
    if (threadIdx.x < st) red[threadIdx.x] += red[threadIdx.x + st];
    __syncthreads();
  }
  float inv = rsqrtf(red[0] / 512.f + 1e-5f);
  for (int c = threadIdx.x; c < 512; c += 256)
    O[(size_t)rowo * 512 + c] = (buf[c] - mu) * inv * g[c] + bb[c];
}

__global__ __launch_bounds__(256) void mean_pool_kernel(
    const float* __restrict__ X, float* __restrict__ O, int n) {
  int i = blockIdx.x * 256 + threadIdx.x;
  if (i >= 16 * 512) return;
  int b = i / 512, c = i % 512;
  float s = 0.f;
  for (int t = 0; t < n; ++t) s += X[((size_t)b * n + t) * 512 + c];
  O[i] = s / n;
}

// ---------------------------------------------------------------------------
// Host side: parameter table + orchestration
// ---------------------------------------------------------------------------
struct LinP  { const float *w, *b; };
struct LnP   { const float *g, *b; };
struct AttnP { LinP qkv, out; };
struct FfnP  { LinP l1, l2; };
struct EncP  { AttnP attn; FfnP ffn; LnP ln1, ln2; const float* cw; const float* cb; LnP cln; };
struct DecP  { AttnP self_a, cross; FfnP ffn; LnP ln1, ln2, ln3; };
struct Model {
  const float* x;
  LinP inp; LnP inp_ln;
  EncP enc[4]; DecP dec[2];
  LinP out1, out2;
};

static void parse_model(void* const* d_in, const int* in_sizes, Model& m) {
  int i = 0;
  auto nx = [&]() { return (const float*)d_in[i++]; };
  if (in_sizes[0] == 16 * 1024 * 32) {
    // dict insertion order: x, params{inp, inp_ln, enc, dec, out1, out2}
    m.x = nx();
    m.inp.w = nx(); m.inp.b = nx();
    m.inp_ln.g = nx(); m.inp_ln.b = nx();
    for (int l = 0; l < 4; ++l) {
      m.enc[l].attn.qkv.w = nx(); m.enc[l].attn.qkv.b = nx();
      m.enc[l].attn.out.w = nx(); m.enc[l].attn.out.b = nx();
      m.enc[l].ffn.l1.w = nx();  m.enc[l].ffn.l1.b = nx();
      m.enc[l].ffn.l2.w = nx();  m.enc[l].ffn.l2.b = nx();
      m.enc[l].ln1.g = nx(); m.enc[l].ln1.b = nx();
      m.enc[l].ln2.g = nx(); m.enc[l].ln2.b = nx();
      if (l < 3) {
        m.enc[l].cw = nx(); m.enc[l].cb = nx();
        m.enc[l].cln.g = nx(); m.enc[l].cln.b = nx();
      }
    }
    for (int l = 0; l < 2; ++l) {
      m.dec[l].self_a.qkv.w = nx(); m.dec[l].self_a.qkv.b = nx();
      m.dec[l].self_a.out.w = nx(); m.dec[l].self_a.out.b = nx();
      m.dec[l].cross.qkv.w = nx();  m.dec[l].cross.qkv.b = nx();
      m.dec[l].cross.out.w = nx();  m.dec[l].cross.out.b = nx();
      m.dec[l].ffn.l1.w = nx(); m.dec[l].ffn.l1.b = nx();
      m.dec[l].ffn.l2.w = nx(); m.dec[l].ffn.l2.b = nx();
      m.dec[l].ln1.g = nx(); m.dec[l].ln1.b = nx();
      m.dec[l].ln2.g = nx(); m.dec[l].ln2.b = nx();
      m.dec[l].ln3.g = nx(); m.dec[l].ln3.b = nx();
    }
    m.out1.w = nx(); m.out1.b = nx();
    m.out2.w = nx(); m.out2.b = nx();
  } else {
    // jax sorted-key pytree flatten order
    for (int l = 0; l < 2; ++l) {
      m.dec[l].cross.out.b = nx(); m.dec[l].cross.out.w = nx();
      m.dec[l].cross.qkv.b = nx(); m.dec[l].cross.qkv.w = nx();
      m.dec[l].ffn.l1.b = nx(); m.dec[l].ffn.l1.w = nx();
      m.dec[l].ffn.l2.b = nx(); m.dec[l].ffn.l2.w = nx();
      m.dec[l].ln1.b = nx(); m.dec[l].ln1.g = nx();
      m.dec[l].ln2.b = nx(); m.dec[l].ln2.g = nx();
      m.dec[l].ln3.b = nx(); m.dec[l].ln3.g = nx();
      m.dec[l].self_a.out.b = nx(); m.dec[l].self_a.out.w = nx();
      m.dec[l].self_a.qkv.b = nx(); m.dec[l].self_a.qkv.w = nx();
    }
    for (int l = 0; l < 4; ++l) {
      m.enc[l].attn.out.b = nx(); m.enc[l].attn.out.w = nx();
      m.enc[l].attn.qkv.b = nx(); m.enc[l].attn.qkv.w = nx();
      if (l < 3) {
        m.enc[l].cb = nx();
        m.enc[l].cln.b = nx(); m.enc[l].cln.g = nx();
        m.enc[l].cw = nx();
      }
      m.enc[l].ffn.l1.b = nx(); m.enc[l].ffn.l1.w = nx();
      m.enc[l].ffn.l2.b = nx(); m.enc[l].ffn.l2.w = nx();
      m.enc[l].ln1.b = nx(); m.enc[l].ln1.g = nx();
      m.enc[l].ln2.b = nx(); m.enc[l].ln2.g = nx();
    }
    m.inp.b = nx();    m.inp.w = nx();
    m.inp_ln.b = nx(); m.inp_ln.g = nx();
    m.out1.b = nx();   m.out1.w = nx();
    m.out2.b = nx();   m.out2.w = nx();
    m.x = nx();
  }
}

static void gemm(const float* A, const float* B, const float* bias, float* C,
                 int M, int N, int K, int act, hipStream_t s) {
  dim3 grid((M + GTM - 1) / GTM, (N + GTN - 1) / GTN);
  gemm_bf16_kernel<<<grid, 256, 0, s>>>(A, B, bias, C, M, N, K, act);
}

static void attention(const float* h, const AttnP& p, float* qkv, float* mb,
                      float* tau, float* aout, float* proj, int n,
                      hipStream_t s) {
  gemm(h, p.qkv.w, p.qkv.b, qkv, 16 * n, 1536, 512, 0, s);
  attn_m_kernel<<<n, 256, 0, s>>>(qkv, mb, n);
  topk_tau_kernel<<<128, 256, 0, s>>>(mb, tau, n, n / 4);
  attn_out_kernel<<<n, 256, 0, s>>>(qkv, mb, tau, aout, n);
  gemm(aout, p.out.w, p.out.b, proj, 16 * n, 512, 512, 0, s);
}

extern "C" void kernel_launch(void* const* d_in, const int* in_sizes, int n_in,
                              void* d_out, int out_size, void* d_ws, size_t ws_size,
                              hipStream_t stream) {
  (void)n_in; (void)out_size; (void)ws_size;
  Model m;
  parse_model(d_in, in_sizes, m);

  float* H    = (float*)d_ws;                         // (16,1024,512) max
  float* T1   = H  + (size_t)16 * 1024 * 512;         // (16,1024,512)
  float* T2   = T1 + (size_t)16 * 1024 * 512;         // (16,1024,2048) max
  float* WP   = T2 + (size_t)16 * 1024 * 2048;        // packed conv weight
  float* MB   = WP + (size_t)1536 * 512;              // sparsity measure
  float* TAU  = MB + (size_t)16 * 8 * 1024;           // per-(b,h) threshold
  float* POOL = TAU + 256;                            // (16,512)
  float* HID  = POOL + 16 * 512;                      // (16,256)
  float* OUTF = (float*)d_out;

  gemm(m.x, m.inp.w, m.inp.b, T1, 16 * 1024, 512, 32, 0, stream);
  ln_kernel<<<16 * 1024, 256, 0, stream>>>(T1, nullptr, m.inp_ln.g, m.inp_ln.b, H, 512);

  int n = 1024;
  for (int l = 0; l < 4; ++l) {
    EncP& e = m.enc[l];
    attention(H, e.attn, T2, MB, TAU, T1, T2, n, stream);
    ln_kernel<<<16 * n, 256, 0, stream>>>(H, T2, e.ln1.g, e.ln1.b, H, 512);
    gemm(H, e.ffn.l1.w, e.ffn.l1.b, T2, 16 * n, 2048, 512, 1, stream);
    gemm(T2, e.ffn.l2.w, e.ffn.l2.b, T1, 16 * n, 512, 2048, 0, stream);
    ln_kernel<<<16 * n, 256, 0, stream>>>(H, T1, e.ln2.g, e.ln2.b, H, 512);
    if (l < 3) {
      conv_pack_kernel<<<(1536 * 512 + 255) / 256, 256, 0, stream>>>(e.cw, WP);
      size_t tot = (size_t)16 * n * 1536;
      im2col_kernel<<<(unsigned)((tot + 255) / 256), 256, 0, stream>>>(H, T2, n);
      gemm(T2, WP, e.cb, T1, 16 * n, 512, 1536, 0, stream);
      pool_elu_ln_kernel<<<16 * (n / 2), 256, 0, stream>>>(T1, e.cln.g, e.cln.b, H, n);
      n >>= 1;
    }
  }
  for (int l = 0; l < 2; ++l) {
    DecP& d = m.dec[l];
    attention(H, d.self_a, T2, MB, TAU, T1, T2, n, stream);
    ln_kernel<<<16 * n, 256, 0, stream>>>(H, T2, d.ln1.g, d.ln1.b, H, 512);
    attention(H, d.cross, T2, MB, TAU, T1, T2, n, stream);   // source bug preserved
    ln_kernel<<<16 * n, 256, 0, stream>>>(H, T2, d.ln2.g, d.ln2.b, H, 512);
    gemm(H, d.ffn.l1.w, d.ffn.l1.b, T2, 16 * n, 2048, 512, 1, stream);
    gemm(T2, d.ffn.l2.w, d.ffn.l2.b, T1, 16 * n, 512, 2048, 0, stream);
    ln_kernel<<<16 * n, 256, 0, stream>>>(H, T1, d.ln3.g, d.ln3.b, H, 512);
  }
  mean_pool_kernel<<<(16 * 512 + 255) / 256, 256, 0, stream>>>(H, POOL, n);
  gemm(POOL, m.out1.w, m.out1.b, HID, 16, 256, 512, 1, stream);
  gemm(HID, m.out2.w, m.out2.b, OUTF, 16, 1, 256, 0, stream);
}
